// MultiMessagePassing_32212254720741
// MI455X (gfx1250) — matrix operand order repaired and checked by reading, not compiled, not run on hardware
//
#include <hip/hip_runtime.h>
#include <hip/hip_bf16.h>

// ---------------------------------------------------------------------------
// Problem constants (fixed by the reference)
// ---------------------------------------------------------------------------
constexpr long NN = 100000;   // nodes
constexpr long EE = 1600000;  // edges
constexpr int  DD = 64;       // node feature dim
constexpr int  AA = 64;       // message dim
constexpr int  GG = 64;       // global dim
constexpr int  SS = 4;        // steps
constexpr int  NG = 16;       // graphs

constexpr long ETILES = EE / 16;   // 100000 (exact)
constexpr long NTILES = NN / 16;   // 6250   (exact)
static_assert(ETILES % 8 == 0, "edge grid must be exact");

typedef __attribute__((ext_vector_type(16))) __bf16 v16bf;
typedef __attribute__((ext_vector_type(8)))  float  v8f;

union bfrag_u { v16bf v; uint4 q[2]; };

// ---------------------------------------------------------------------------
// Helpers
// ---------------------------------------------------------------------------
__device__ __forceinline__ int fkey(float v) {
  int i = __float_as_int(v);
  return (i >= 0) ? i : (i ^ 0x7fffffff);   // monotonic float -> signed int
}
__device__ __forceinline__ float funkey(int k) {
  if (k == (int)0x80000000) return 0.0f;    // empty-segment sentinel -> 0
  int bits = (k >= 0) ? k : (k ^ 0x7fffffff);
  return __int_as_float(bits);
}
__device__ __forceinline__ float lrelu(float v) { return v > 0.f ? v : 0.01f * v; }

// A-fragment (16x32 bf16) per CDNA5 layout:
// lane<16 : row M=lane,    elems 0..7 = K k0..k0+7,  elems 8..15 = K k0+16..k0+23
// lane>=16: row M=lane-16, elems 0..7 = K k0+8..+15, elems 8..15 = K k0+24..+31
__device__ __forceinline__ v16bf load_a_f32(const float* __restrict__ row, int k0, int half) {
  v16bf a;
  const float* p1 = row + k0 + half * 8;
  const float* p2 = row + k0 + 16 + half * 8;
#pragma unroll
  for (int j = 0; j < 8; ++j) {
    a[j]     = (__bf16)p1[j];
    a[8 + j] = (__bf16)p2[j];
  }
  return a;
}
__device__ __forceinline__ v16bf load_a_key(const int* __restrict__ row, int k0, int half) {
  v16bf a;
  const int* p1 = row + k0 + half * 8;
  const int* p2 = row + k0 + 16 + half * 8;
#pragma unroll
  for (int j = 0; j < 8; ++j) {
    a[j]     = (__bf16)funkey(p1[j]);
    a[8 + j] = (__bf16)funkey(p2[j]);
  }
  return a;
}

// B-fragment: per (nt,kc) block of 512 bf16 = two 256-element planes.
// Plane h holds, for each lane, elements j = h*8..h*8+7 as 16B chunks with
// lane stride 16B  -> bank-conflict-free ds_load_b128.
__device__ __forceinline__ v16bf load_b_frag(const __bf16* base, int lane) {
  bfrag_u u;
  u.q[0] = *(const uint4*)(base + lane * 8);          // plane 0
  u.q[1] = *(const uint4*)(base + 256 + lane * 8);    // plane 1
  return u.v;
}

// ---------------------------------------------------------------------------
// Weight fragment pre-pack into split-plane B layout:
// idx = ((((s*NT+nt)*KC+kc)*2 + h)*32 + lane)*8 + j2
// element (lane, j=h*8+j2)  <-  W[s][kc*32 + h*16 + (lane>>4)*8 + j2][nt*16 + (lane&15)]
// ---------------------------------------------------------------------------
__global__ __launch_bounds__(256) void prep_frag_kernel(
    const float* __restrict__ W, __bf16* __restrict__ out, int K, int Nn, int KC, int NT) {
  long idx = (long)blockIdx.x * 256 + threadIdx.x;
  long total = (long)SS * NT * KC * 512;
  if (idx >= total) return;
  int j2 = idx & 7;
  long t = idx >> 3;
  int lane = t & 31; t >>= 5;
  int h  = (int)(t & 1); t >>= 1;
  int kc = (int)(t % KC); t /= KC;
  int nt = (int)(t % NT); t /= NT;
  int s  = (int)t;
  int k = kc * 32 + h * 16 + (lane >> 4) * 8 + j2;
  int n = nt * 16 + (lane & 15);
  out[idx] = (__bf16)W[((long)s * K + k) * Nn + n];
}

// ---------------------------------------------------------------------------
// Per-step scratch init
// ---------------------------------------------------------------------------
__global__ __launch_bounds__(256) void init_step_kernel(
    int* __restrict__ agg, int* __restrict__ gmax, float* __restrict__ denom,
    float* __restrict__ pooled) {
  long i = (long)blockIdx.x * 256 + threadIdx.x;
  if (i < NN * 64L) agg[i] = (int)0x80000000;
  if (i < NG) { gmax[i] = (int)0x80000000; denom[i] = 0.f; }
  if (i < NG * 64) pooled[i] = 0.f;
}

// ---------------------------------------------------------------------------
// Kernel 1: fused gather -> WMMA edge GEMM -> lrelu -> scatter atomic-max
// one wave per 16-edge tile, 8 waves / block; weights staged in LDS
// ---------------------------------------------------------------------------
__global__ __launch_bounds__(256) void edge_msg_kernel(
    const float* __restrict__ x, const int* __restrict__ src, const int* __restrict__ dst,
    const __bf16* __restrict__ fragWm, const float* __restrict__ bm,
    int* __restrict__ agg_key) {
  __shared__ __bf16 sB[4 * 2 * 512];  // 8 KB
  for (int i = threadIdx.x; i < 4 * 2 * 512 / 8; i += 256)
    ((uint4*)sB)[i] = ((const uint4*)fragWm)[i];
  __syncthreads();

  const int lane = threadIdx.x & 31;
  const int wave = threadIdx.x >> 5;
  const long t = (long)blockIdx.x * 8 + wave;   // exact grid: no guard needed
  const int half = lane >> 4;
  const int m    = lane & 15;
  const long e_a = t * 16 + m;
  const float* arow = x + (long)src[e_a] * DD;
  __builtin_prefetch(src + e_a + 8192, 0, 0);
  __builtin_prefetch(dst + e_a + 8192, 0, 0);

  v8f c[4] = {};
#pragma unroll
  for (int kc = 0; kc < 2; ++kc) {
    v16bf a = load_a_f32(arow, kc * 32, half);
#pragma unroll
    for (int nt = 0; nt < 4; ++nt) {
      v16bf b = load_b_frag(sB + (nt * 2 + kc) * 512, lane);
      c[nt] = __builtin_amdgcn_wmma_f32_16x16x32_bf16(false, a, false, b, (short)0,
                                                      c[nt], false, false);
    }
  }
  int drow[8];
#pragma unroll
  for (int r = 0; r < 8; ++r) drow[r] = dst[t * 16 + r + half * 8];
  const int cn = lane & 15;
#pragma unroll
  for (int nt = 0; nt < 4; ++nt) {
    const int col = nt * 16 + cn;
    const float bias = bm[col];
#pragma unroll
    for (int r = 0; r < 8; ++r) {
      float v = lrelu(c[nt][r] + bias);
      atomicMax(&agg_key[(long)drow[r] * AA + col], fkey(v));
    }
  }
}

// ---------------------------------------------------------------------------
// Kernel 2: node update  x' = lrelu([x, xg[batch], agg] @ Wa + ba) + x
// K = 192 (3 sources x 64), one wave per 16-node tile; weights staged in LDS
// ---------------------------------------------------------------------------
__global__ __launch_bounds__(256) void node_update_kernel(
    const float* __restrict__ x, const float* __restrict__ xg,
    const int* __restrict__ batch, const int* __restrict__ agg_key,
    const __bf16* __restrict__ fragWa, const float* __restrict__ ba,
    float* __restrict__ x_out) {
  __shared__ __bf16 sB[4 * 6 * 512];  // 24 KB
  for (int i = threadIdx.x; i < 4 * 6 * 512 / 8; i += 256)
    ((uint4*)sB)[i] = ((const uint4*)fragWa)[i];
  __syncthreads();

  const int lane = threadIdx.x & 31;
  const int wave = threadIdx.x >> 5;
  const long t = (long)blockIdx.x * 8 + wave;
  if (t >= NTILES) return;  // wave-uniform, after barrier
  const int half = lane >> 4;
  const int m    = lane & 15;
  const long n_a = t * 16 + m;
  const float* xrow = x + n_a * DD;
  const float* grow = xg + (long)batch[n_a] * GG;
  const int*   krow = agg_key + n_a * AA;

  v8f c[4] = {};
#pragma unroll
  for (int kc = 0; kc < 6; ++kc) {
    v16bf a;
    if (kc < 2)      a = load_a_f32(xrow, kc * 32, half);
    else if (kc < 4) a = load_a_f32(grow, (kc - 2) * 32, half);
    else             a = load_a_key(krow, (kc - 4) * 32, half);
#pragma unroll
    for (int nt = 0; nt < 4; ++nt) {
      v16bf b = load_b_frag(sB + (nt * 6 + kc) * 512, lane);
      c[nt] = __builtin_amdgcn_wmma_f32_16x16x32_bf16(false, a, false, b, (short)0,
                                                      c[nt], false, false);
    }
  }
  const int cn = lane & 15;
#pragma unroll
  for (int nt = 0; nt < 4; ++nt) {
    const int col = nt * 16 + cn;
    const float bias = ba[col];
#pragma unroll
    for (int r = 0; r < 8; ++r) {
      const long n = t * 16 + r + half * 8;
      x_out[n * DD + col] = lrelu(c[nt][r] + bias) + x[n * DD + col];
    }
  }
}

// ---------------------------------------------------------------------------
// Kernel 3: gate = x' @ Wgate + bgate, segment max per graph (LDS hierarchy)
// ---------------------------------------------------------------------------
__global__ __launch_bounds__(256) void gate_kernel(
    const float* __restrict__ x, const float* __restrict__ Wg, const float* __restrict__ bg,
    const int* __restrict__ batch, float* __restrict__ gate, int* __restrict__ gmax_key) {
  __shared__ int lmax[NG];
  const int tid = threadIdx.x;
  if (tid < NG) lmax[tid] = (int)0x80000000;
  __syncthreads();
  const long n = (long)blockIdx.x * 256 + tid;
  if (n < NN) {
    const float* row = x + n * DD;
    float acc = bg[0];
#pragma unroll
    for (int k = 0; k < DD; ++k) acc += row[k] * Wg[k];
    gate[n] = acc;
    atomicMax(&lmax[batch[n]], fkey(acc));
  }
  __syncthreads();
  if (tid < NG && lmax[tid] != (int)0x80000000) atomicMax(&gmax_key[tid], lmax[tid]);
}

// ---------------------------------------------------------------------------
// Kernel 4: e = exp(gate - gmax[batch]); denom = segment_sum(e)
// ---------------------------------------------------------------------------
__global__ __launch_bounds__(256) void exp_kernel(
    const float* __restrict__ gate, const int* __restrict__ gmax_key,
    const int* __restrict__ batch, float* __restrict__ ebuf, float* __restrict__ denom) {
  __shared__ float lsum[NG];
  const int tid = threadIdx.x;
  if (tid < NG) lsum[tid] = 0.f;
  __syncthreads();
  const long n = (long)blockIdx.x * 256 + tid;
  if (n < NN) {
    const int g = batch[n];
    const float ev = __expf(gate[n] - funkey(gmax_key[g]));
    ebuf[n] = ev;
    atomicAdd(&lsum[g], ev);
  }
  __syncthreads();
  if (tid < NG && lsum[tid] != 0.f) atomicAdd(&denom[tid], lsum[tid]);
}

// ---------------------------------------------------------------------------
// Kernel 5: feat = lrelu(x' @ Wfeat + bfeat); pooled += alpha * feat (LDS hierarchy)
// ---------------------------------------------------------------------------
__global__ __launch_bounds__(256) void feat_pool_kernel(
    const float* __restrict__ x, const float* __restrict__ ebuf,
    const float* __restrict__ denom, const int* __restrict__ batch,
    const __bf16* __restrict__ fragWf, const float* __restrict__ bfeat,
    float* __restrict__ pooled) {
  __shared__ __bf16 sB[4 * 2 * 512];  // 8 KB
  __shared__ float pl[NG * 64];       // 4 KB
  for (int i = threadIdx.x; i < 4 * 2 * 512 / 8; i += 256)
    ((uint4*)sB)[i] = ((const uint4*)fragWf)[i];
  for (int i = threadIdx.x; i < NG * 64; i += 256) pl[i] = 0.f;
  __syncthreads();

  const int lane = threadIdx.x & 31;
  const int wave = threadIdx.x >> 5;
  const long t = (long)blockIdx.x * 8 + wave;
  const bool active = (t < NTILES);  // wave-uniform
  if (active) {
    const int half = lane >> 4;
    const int m    = lane & 15;
    const float* xrow = x + (t * 16 + m) * DD;
    v8f c[4] = {};
#pragma unroll
    for (int kc = 0; kc < 2; ++kc) {
      v16bf a = load_a_f32(xrow, kc * 32, half);
#pragma unroll
      for (int nt = 0; nt < 4; ++nt) {
        v16bf b = load_b_frag(sB + (nt * 2 + kc) * 512, lane);
        c[nt] = __builtin_amdgcn_wmma_f32_16x16x32_bf16(false, a, false, b, (short)0,
                                                        c[nt], false, false);
      }
    }
    const int cn = lane & 15;
#pragma unroll
    for (int nt = 0; nt < 4; ++nt) {
      const int col = nt * 16 + cn;
      const float bias = bfeat[col];
#pragma unroll
      for (int r = 0; r < 8; ++r) {
        const long n = t * 16 + r + half * 8;
        const int g = batch[n];
        const float alpha = ebuf[n] / denom[g];
        atomicAdd(&pl[g * 64 + col], alpha * lrelu(c[nt][r] + bias));
      }
    }
  }
  __syncthreads();
  for (int i = threadIdx.x; i < NG * 64; i += 256) {
    const float v = pl[i];
    if (v != 0.f) atomicAdd(&pooled[i], v);
  }
}

// ---------------------------------------------------------------------------
// Kernel 6: global update  xg' = lrelu([pooled, xg] @ Wt + bt) + xg
// M=16 (= NG), K=128, one wave, one block
// ---------------------------------------------------------------------------
__global__ __launch_bounds__(32) void global_update_kernel(
    const float* __restrict__ pooled, const float* __restrict__ g_in,
    const __bf16* __restrict__ fragWt, const float* __restrict__ bt,
    float* __restrict__ g_out) {
  const int lane = threadIdx.x;
  const int half = lane >> 4;
  const int m    = lane & 15;
  const float* prow = pooled + m * GG;
  const float* grow = g_in + m * GG;

  v8f c[4] = {};
#pragma unroll
  for (int kc = 0; kc < 4; ++kc) {
    v16bf a = (kc < 2) ? load_a_f32(prow, kc * 32, half)
                       : load_a_f32(grow, (kc - 2) * 32, half);
#pragma unroll
    for (int nt = 0; nt < 4; ++nt) {
      v16bf b = load_b_frag(fragWt + (nt * 4 + kc) * 512, lane);
      c[nt] = __builtin_amdgcn_wmma_f32_16x16x32_bf16(false, a, false, b, (short)0,
                                                      c[nt], false, false);
    }
  }
  const int cn = lane & 15;
#pragma unroll
  for (int nt = 0; nt < 4; ++nt) {
    const int col = nt * 16 + cn;
    const float bias = bt[col];
#pragma unroll
    for (int r = 0; r < 8; ++r) {
      const int g = r + half * 8;
      g_out[g * GG + col] = lrelu(c[nt][r] + bias) + g_in[g * GG + col];
    }
  }
}

// ---------------------------------------------------------------------------
// Final: pack (x, x_global) into d_out
// ---------------------------------------------------------------------------
__global__ __launch_bounds__(256) void pack_out_kernel(
    const float* __restrict__ x, const float* __restrict__ xg, float* __restrict__ out) {
  long i = (long)blockIdx.x * 256 + threadIdx.x;
  const long nx = NN * 64L;
  if (i < nx) out[i] = x[i];
  else if (i < nx + NG * 64L) out[i] = xg[i - nx];
}

// ---------------------------------------------------------------------------
// Host launcher
// ---------------------------------------------------------------------------
extern "C" void kernel_launch(void* const* d_in, const int* in_sizes, int n_in,
                              void* d_out, int out_size, void* d_ws, size_t ws_size,
                              hipStream_t stream) {
  const float* x_in   = (const float*)d_in[0];
  const float* g_in   = (const float*)d_in[1];
  const int*   eidx   = (const int*)d_in[3];
  const int*   batch  = (const int*)d_in[4];
  const float* Wm     = (const float*)d_in[6];
  const float* bm     = (const float*)d_in[7];
  const float* Wa     = (const float*)d_in[8];
  const float* ba     = (const float*)d_in[9];
  const float* Wgate  = (const float*)d_in[10];
  const float* bgate  = (const float*)d_in[11];
  const float* Wfeat  = (const float*)d_in[12];
  const float* bfeat  = (const float*)d_in[13];
  const float* Wt     = (const float*)d_in[14];
  const float* bt     = (const float*)d_in[15];
  const int* src = eidx;
  const int* dst = eidx + EE;

  // ---- carve workspace (256B aligned segments) ----
  char* ws = (char*)d_ws;
  size_t off = 0;
  auto carve = [&](size_t bytes) -> void* {
    void* p = ws + off;
    off += (bytes + 255) & ~(size_t)255;
    return p;
  };
  float*  xbuf0  = (float*)carve(NN * 64 * 4);
  float*  xbuf1  = (float*)carve(NN * 64 * 4);
  float*  gbuf0  = (float*)carve(NG * 64 * 4);
  float*  gbuf1  = (float*)carve(NG * 64 * 4);
  int*    aggk   = (int*)carve(NN * 64 * 4);
  float*  gateb  = (float*)carve(NN * 4);
  float*  ebuf   = (float*)carve(NN * 4);
  int*    gmaxk  = (int*)carve(NG * 4);
  float*  denom  = (float*)carve(NG * 4);
  float*  pooled = (float*)carve(NG * 64 * 4);
  __bf16* fWm    = (__bf16*)carve((size_t)SS * 4 * 2 * 512 * 2);  // [s][nt][kc][plane][lane][8]
  __bf16* fWa    = (__bf16*)carve((size_t)SS * 4 * 6 * 512 * 2);  // K=192 -> kc6
  __bf16* fWf    = (__bf16*)carve((size_t)SS * 4 * 2 * 512 * 2);
  __bf16* fWt    = (__bf16*)carve((size_t)SS * 4 * 4 * 512 * 2);  // K=128 -> kc4

  // ---- pre-pack all step weights into WMMA B-fragment layout ----
  {
    long tWm = (long)SS * 4 * 2 * 512;  // 16384
    long tWa = (long)SS * 4 * 6 * 512;  // 49152
    long tWf = (long)SS * 4 * 2 * 512;
    long tWt = (long)SS * 4 * 4 * 512;  // 32768
    prep_frag_kernel<<<(int)((tWm + 255) / 256), 256, 0, stream>>>(Wm, fWm, 64, 64, 2, 4);
    prep_frag_kernel<<<(int)((tWa + 255) / 256), 256, 0, stream>>>(Wa, fWa, 192, 64, 6, 4);
    prep_frag_kernel<<<(int)((tWf + 255) / 256), 256, 0, stream>>>(Wfeat, fWf, 64, 64, 2, 4);
    prep_frag_kernel<<<(int)((tWt + 255) / 256), 256, 0, stream>>>(Wt, fWt, 128, 64, 4, 4);
  }

  const int initBlocks = (int)((NN * 64L + 255) / 256);         // 25000
  const int edgeBlocks = (int)(ETILES / 8);                     // 12500 exact
  const int nodeBlocks = (int)((NTILES + 7) / 8);               // 782
  const int rowBlocks  = (int)((NN + 255) / 256);               // 391
  const int packBlocks = (int)((NN * 64L + NG * 64L + 255) / 256);

  const float* cur_x = x_in;
  const float* cur_g = g_in;
  for (int s = 0; s < SS; ++s) {
    float* out_x = (s & 1) ? xbuf1 : xbuf0;
    float* out_g = (s & 1) ? gbuf1 : gbuf0;

    init_step_kernel<<<initBlocks, 256, 0, stream>>>(aggk, gmaxk, denom, pooled);

    edge_msg_kernel<<<edgeBlocks, 256, 0, stream>>>(
        cur_x, src, dst, fWm + (long)s * 4 * 2 * 512, bm + s * 64, aggk);

    node_update_kernel<<<nodeBlocks, 256, 0, stream>>>(
        cur_x, cur_g, batch, aggk, fWa + (long)s * 4 * 6 * 512, ba + s * 64, out_x);

    gate_kernel<<<rowBlocks, 256, 0, stream>>>(
        out_x, Wgate + s * 64, bgate + s, batch, gateb, gmaxk);

    exp_kernel<<<rowBlocks, 256, 0, stream>>>(gateb, gmaxk, batch, ebuf, denom);

    feat_pool_kernel<<<nodeBlocks, 256, 0, stream>>>(
        out_x, ebuf, denom, batch, fWf + (long)s * 4 * 2 * 512, bfeat + s * 64, pooled);

    global_update_kernel<<<1, 32, 0, stream>>>(
        pooled, cur_g, fWt + (long)s * 4 * 4 * 512, bt + s * 64, out_g);

    cur_x = out_x;
    cur_g = out_g;
  }

  pack_out_kernel<<<packBlocks, 256, 0, stream>>>(cur_x, cur_g, (float*)d_out);
}
